// TransformerBlock_36653250904280
// MI455X (gfx1250) — compile-verified, matrix operand order checked by reading, not run on hardware
//
#include <hip/hip_runtime.h>
#include <hip/hip_bf16.h>
#include <math.h>

// ---------------------------------------------------------------------------
// Types for WMMA fragments (gfx1250, wave32)
// ---------------------------------------------------------------------------
typedef _Float16 h16 __attribute__((ext_vector_type(16)));
typedef _Float16 h8  __attribute__((ext_vector_type(8)));
typedef float    f8v __attribute__((ext_vector_type(8)));

#define M_TOT   131072      // B*H*W = 8*128*128
#define DIM     192
#define HEADS   6
#define HD      32          // head dim
#define NQKV    576
#define NIN     1024        // HIDDEN*2 = 1020 padded to 1024
#define KOUT    512         // HIDDEN = 510 padded to 512

// Assemble a 16-half WMMA operand fragment from two contiguous 16B chunks.
// Layout (ISA 7.12.2, 16-bit A 16x32): lane(g=lane/16, m=lane&15) holds
// K = {g*8..g*8+7} in v0..3 and K = {16+g*8..+7} in v4..7 -> 2x b128 loads.
static __device__ __forceinline__ h16 frag2(const _Float16* p0, const _Float16* p1) {
    h8 lo = *(const h8*)p0;
    h8 hi = *(const h8*)p1;
    h16 f;
#pragma unroll
    for (int i = 0; i < 8; ++i) { f[i] = lo[i]; f[i + 8] = hi[i]; }
    return f;
}

static __device__ __forceinline__ f8v wmma_f16(h16 a, h16 b, f8v c) {
    return __builtin_amdgcn_wmma_f32_16x16x32_f16(false, a, false, b,
                                                  (short)0, c, false, false);
}

// ---------------------------------------------------------------------------
// K1: cast/transpose all weights to f16 [N][K] layout (B-operand layout)
// ---------------------------------------------------------------------------
__global__ void convert_weights(const float* __restrict__ wq,   // [192][576]
                                const float* __restrict__ wp,   // [192][192]
                                const float* __restrict__ wi,   // [192][1020]
                                const float* __restrict__ wo,   // [510][192]
                                _Float16* __restrict__ tq,      // [576][192]
                                _Float16* __restrict__ tp,      // [192][192]
                                _Float16* __restrict__ ti,      // [1024][192]
                                _Float16* __restrict__ to_) {   // [192][512]
    unsigned i = blockIdx.x * 256u + threadIdx.x;
    if (i < 110592u) {                       // wqkv: [576][192]
        unsigned n = i / 192u, k = i % 192u;
        tq[i] = (_Float16)wq[k * 576u + n];
    } else if (i < 147456u) {                // wproj: [192][192]
        unsigned j = i - 110592u, n = j / 192u, k = j % 192u;
        tp[j] = (_Float16)wp[k * 192u + n];
    } else if (i < 344064u) {                // w_in: [1024][192], rows>=1020 zero
        unsigned j = i - 147456u, n = j / 192u, k = j % 192u;
        ti[j] = (n < 1020u) ? (_Float16)wi[k * 1020u + n] : (_Float16)0.0f;
    } else {                                 // w_out: [192][512], cols>=510 zero
        unsigned j = i - 344064u, n = j / 512u, k = j % 512u;
        to_[j] = (k < 510u) ? (_Float16)wo[k * 192u + n] : (_Float16)0.0f;
    }
}

// ---------------------------------------------------------------------------
// K2/K9: LayerNorm over 192 channels, f32 in -> f16 out. One wave per row.
// ---------------------------------------------------------------------------
__global__ void __launch_bounds__(256)
ln_kernel(const float* __restrict__ X, const float* __restrict__ gam,
          const float* __restrict__ bet, _Float16* __restrict__ Y) {
    int w = threadIdx.x >> 5, lane = threadIdx.x & 31;
    size_t row = (size_t)blockIdx.x * 8 + w;
    const float* xr = X + row * DIM;
    float v[6], s = 0.f, q = 0.f;
#pragma unroll
    for (int j = 0; j < 6; ++j) {
        v[j] = xr[j * 32 + lane];
        s += v[j]; q += v[j] * v[j];
    }
#pragma unroll
    for (int o = 16; o > 0; o >>= 1) {
        s += __shfl_xor(s, o, 32);
        q += __shfl_xor(q, o, 32);
    }
    float mu  = s * (1.0f / 192.0f);
    float var = q * (1.0f / 192.0f) - mu * mu;
    float rs  = rsqrtf(var + 1e-3f);
    _Float16* yr = Y + row * DIM;
#pragma unroll
    for (int j = 0; j < 6; ++j) {
        int c = j * 32 + lane;
        yr[c] = (_Float16)((v[j] - mu) * rs * gam[c] + bet[c]);
    }
}

// ---------------------------------------------------------------------------
// Generic WMMA GEMM:  Out[M][ldo] = A[M][K](f16) x B[N][K](f16)^T  (+ Res)
// Block = 256 thr = 8 waves (4 M-waves x 2 N-waves); WG tile 128M x 64N.
// Each wave: 2x2 tiles of 16x16 -> 4 wmma per k-step. B tile staged in LDS.
// MODE 0: f16 out, no residual.  MODE 1: f32 out + f32 residual.
// ---------------------------------------------------------------------------
template <int MODE>
__global__ void __launch_bounds__(256)
gemm_kernel(const _Float16* __restrict__ A, const _Float16* __restrict__ B,
            const float* __restrict__ Res, void* __restrict__ Out,
            int K, int ldo) {
    __shared__ __align__(16) _Float16 ldsB[64 * 32];
    int t = threadIdx.x;
    int w = t >> 5, lane = t & 31, g = lane >> 4, ml = lane & 15;
    int wm = w & 3, wn = w >> 2;
    int n0 = blockIdx.x * 64, m0 = blockIdx.y * 128;

    f8v acc[2][2] = {};
    for (int k0 = 0; k0 < K; k0 += 32) {
        __syncthreads();
        {   // cooperative B tile load: 64 rows x 32 halves
            int nl = t >> 2, seg = t & 3;
            ((h8*)ldsB)[t] = *(const h8*)(B + (size_t)(n0 + nl) * K + k0 + seg * 8);
        }
        __syncthreads();
        h16 a[2], b[2];
#pragma unroll
        for (int tm = 0; tm < 2; ++tm) {
            const _Float16* pa =
                A + (size_t)(m0 + wm * 32 + tm * 16 + ml) * K + k0 + g * 8;
            if (k0 + 32 < K) __builtin_prefetch(pa + 32, 0, 0);
            a[tm] = frag2(pa, pa + 16);
        }
#pragma unroll
        for (int tn = 0; tn < 2; ++tn) {
            const _Float16* pb = ldsB + (wn * 32 + tn * 16 + ml) * 32 + g * 8;
            b[tn] = frag2(pb, pb + 16);
        }
#pragma unroll
        for (int tm = 0; tm < 2; ++tm)
#pragma unroll
            for (int tn = 0; tn < 2; ++tn)
                acc[tm][tn] = wmma_f16(a[tm], b[tn], acc[tm][tn]);
    }
#pragma unroll
    for (int tm = 0; tm < 2; ++tm)
#pragma unroll
        for (int tn = 0; tn < 2; ++tn)
#pragma unroll
            for (int r = 0; r < 8; ++r) {
                int m = m0 + wm * 32 + tm * 16 + r + 8 * g;
                int n = n0 + wn * 32 + tn * 16 + ml;
                float val = acc[tm][tn][r];
                if (MODE == 0)
                    ((_Float16*)Out)[(size_t)m * ldo + n] = (_Float16)val;
                else
                    ((float*)Out)[(size_t)m * ldo + n] =
                        val + Res[(size_t)m * ldo + n];
            }
}

// ---------------------------------------------------------------------------
// K4: depthwise 3x3 over qkv (576 ch) + q/k per-pixel normalization,
//     scatter into head-major f16 buffers q_h/k_h/v_h [48][16384][32].
// ---------------------------------------------------------------------------
__global__ void dwconv_qkv_kernel(const _Float16* __restrict__ qkv,
                                  const float* __restrict__ wdw,   // [3][3][1][576]
                                  const float* __restrict__ mq,
                                  const float* __restrict__ vq,
                                  const float* __restrict__ mk,
                                  const float* __restrict__ vk,
                                  _Float16* __restrict__ qh,
                                  _Float16* __restrict__ kh,
                                  _Float16* __restrict__ vh) {
    unsigned idx = blockIdx.x * 256u + threadIdx.x;
    int c = (int)(idx % 576u);
    unsigned m = idx / 576u;
    int b = (int)(m >> 14), n = (int)(m & 16383u);
    int y = n >> 7, xp = n & 127;
    float acc = 0.0f;
#pragma unroll
    for (int dy = -1; dy <= 1; ++dy) {
        int yy = y + dy;
        if ((unsigned)yy >= 128u) continue;
#pragma unroll
        for (int dx = -1; dx <= 1; ++dx) {
            int xx = xp + dx;
            if ((unsigned)xx >= 128u) continue;
            size_t src = ((size_t)((b << 14) + (yy << 7) + xx)) * 576 + c;
            acc += (float)qkv[src] * wdw[((dy + 1) * 3 + (dx + 1)) * 576 + c];
        }
    }
    if (c < 192) {
        int head = c >> 5, d = c & 31;
        float v = (acc - mq[n]) * rsqrtf(vq[n]);
        qh[(((size_t)b * HEADS + head) * 16384 + n) * 32 + d] = (_Float16)v;
    } else if (c < 384) {
        int c2 = c - 192, head = c2 >> 5, d = c2 & 31;
        float v = (acc - mk[n]) * rsqrtf(vk[n]);
        kh[(((size_t)b * HEADS + head) * 16384 + n) * 32 + d] = (_Float16)v;
    } else {
        int c2 = c - 384, head = c2 >> 5, d = c2 & 31;
        vh[(((size_t)b * HEADS + head) * 16384 + n) * 32 + d] = (_Float16)acc;
    }
}

// ---------------------------------------------------------------------------
// K5: attention partial GEMM  attn[c,d] = sum_n k[n,c]*q[n,d]  (per b,head)
// grid(8 slices, 48 bh) x 256 thr. Split-K over n, LDS-transposed operands,
// cross-wave reduction in LDS -> deterministic partials (no atomics).
// ---------------------------------------------------------------------------
struct __align__(16) SM5 {
    union {
        _Float16 tr[2][32 * 264];   // transposed k / q chunks (padded rows)
        float    red[8 * 1024];     // per-wave 32x32 partials
    };
};

__global__ void __launch_bounds__(256)
attn_kernel(const _Float16* __restrict__ qh, const _Float16* __restrict__ kh,
            float* __restrict__ part) {
    __shared__ SM5 sm;
    int bh = blockIdx.y, slice = blockIdx.x;
    int t = threadIdx.x;
    int w = t >> 5, lane = t & 31, g = lane >> 4, ml = lane & 15;
    const _Float16* Q  = qh + (size_t)bh * 16384 * 32;
    const _Float16* Kp = kh + (size_t)bh * 16384 * 32;

    f8v acc[2][2] = {};
    for (int chunk = 0; chunk < 8; ++chunk) {
        __syncthreads();
        {   // load 256 rows of q/k (32 halves each) and transpose into LDS
            int n = slice * 2048 + chunk * 256 + t;
            const h8* pk = (const h8*)(Kp + (size_t)n * 32);
            const h8* pq = (const h8*)(Q  + (size_t)n * 32);
#pragma unroll
            for (int seg = 0; seg < 4; ++seg) {
                h8 kv = pk[seg], qv = pq[seg];
#pragma unroll
                for (int i = 0; i < 8; ++i) {
                    sm.tr[0][(seg * 8 + i) * 264 + t] = kv[i];
                    sm.tr[1][(seg * 8 + i) * 264 + t] = qv[i];
                }
            }
        }
        __syncthreads();
        h16 a[2], b[2];
#pragma unroll
        for (int tm = 0; tm < 2; ++tm) {   // A = k^T rows = channel c
            const _Float16* p = &sm.tr[0][(tm * 16 + ml) * 264 + w * 32 + g * 8];
            a[tm] = frag2(p, p + 16);
        }
#pragma unroll
        for (int tn = 0; tn < 2; ++tn) {   // B^T rows = channel d (from q)
            const _Float16* p = &sm.tr[1][(tn * 16 + ml) * 264 + w * 32 + g * 8];
            b[tn] = frag2(p, p + 16);
        }
#pragma unroll
        for (int tm = 0; tm < 2; ++tm)
#pragma unroll
            for (int tn = 0; tn < 2; ++tn)
                acc[tm][tn] = wmma_f16(a[tm], b[tn], acc[tm][tn]);
    }
    __syncthreads();
#pragma unroll
    for (int tm = 0; tm < 2; ++tm)
#pragma unroll
        for (int tn = 0; tn < 2; ++tn)
#pragma unroll
            for (int r = 0; r < 8; ++r)
                sm.red[w * 1024 + (tm * 16 + r + 8 * g) * 32 + tn * 16 + ml] =
                    acc[tm][tn][r];
    __syncthreads();
#pragma unroll
    for (int o = 0; o < 4; ++o) {
        int idx = t * 4 + o;
        float s = 0.f;
#pragma unroll
        for (int ww = 0; ww < 8; ++ww) s += sm.red[ww * 1024 + idx];
        part[((size_t)bh * 8 + slice) * 1024 + idx] = s;
    }
}

// ---------------------------------------------------------------------------
// K5b: reduce split-K slices, *temp, softmax over c (axis=-2), write
//      attnT[bh][d][c] f16 (B-operand layout for the V*attn GEMM).
// ---------------------------------------------------------------------------
__global__ void softmax_kernel(const float* __restrict__ part,
                               const float* __restrict__ temp,
                               _Float16* __restrict__ attnT) {
    int bh = blockIdx.x, d = threadIdx.x;          // 32 threads
    float tv = temp[bh % HEADS];
    float a[32], mx = -1e30f;
#pragma unroll
    for (int c = 0; c < 32; ++c) {
        float s = 0.f;
        for (int sl = 0; sl < 8; ++sl)
            s += part[((size_t)bh * 8 + sl) * 1024 + c * 32 + d];
        s *= tv;
        a[c] = s;
        mx = fmaxf(mx, s);
    }
    float den = 0.f;
#pragma unroll
    for (int c = 0; c < 32; ++c) { a[c] = expf(a[c] - mx); den += a[c]; }
    float inv = 1.0f / den;
#pragma unroll
    for (int c = 0; c < 32; ++c)
        attnT[(size_t)bh * 1024 + d * 32 + c] = (_Float16)(a[c] * inv);
}

// ---------------------------------------------------------------------------
// K6: out[n,d] = sum_c v[n,c] * attn[c,d]  per (b,head); K=32 single step.
// Writes head-interleaved out_heads f16 [M][192]. grid(64 mchunks, 48 bh).
// ---------------------------------------------------------------------------
__global__ void __launch_bounds__(256)
gemm_av_kernel(const _Float16* __restrict__ vh,
               const _Float16* __restrict__ attnT,
               _Float16* __restrict__ outH) {
    int bh = blockIdx.y;
    int t = threadIdx.x;
    int w = t >> 5, lane = t & 31, g = lane >> 4, ml = lane & 15;
    int m0 = blockIdx.x * 256 + w * 32;
    const _Float16* V  = vh + (size_t)bh * 16384 * 32;
    const _Float16* Bt = attnT + (size_t)bh * 1024;

    h16 a[2], b[2];
#pragma unroll
    for (int tm = 0; tm < 2; ++tm) {
        const _Float16* p = V + (size_t)(m0 + tm * 16 + ml) * 32 + g * 8;
        a[tm] = frag2(p, p + 16);
    }
#pragma unroll
    for (int tn = 0; tn < 2; ++tn) {
        const _Float16* p = Bt + (tn * 16 + ml) * 32 + g * 8;
        b[tn] = frag2(p, p + 16);
    }
    f8v acc[2][2] = {};
#pragma unroll
    for (int tm = 0; tm < 2; ++tm)
#pragma unroll
        for (int tn = 0; tn < 2; ++tn)
            acc[tm][tn] = wmma_f16(a[tm], b[tn], acc[tm][tn]);

    int bimg = bh / HEADS, head = bh % HEADS;
#pragma unroll
    for (int tm = 0; tm < 2; ++tm)
#pragma unroll
        for (int tn = 0; tn < 2; ++tn)
#pragma unroll
            for (int r = 0; r < 8; ++r) {
                int mm = m0 + tm * 16 + r + 8 * g;
                int nn = tn * 16 + ml;
                outH[((size_t)bimg * 16384 + mm) * DIM + head * 32 + nn] =
                    (_Float16)acc[tm][tn][r];
            }
}

// ---------------------------------------------------------------------------
// K10: depthwise 3x3 over 1020 channels + gelu(u1)*u2 gate -> u f16 [M][512]
// ---------------------------------------------------------------------------
__global__ void dwconv_ffn_kernel(const _Float16* __restrict__ upre, // [M][1024]
                                  const float* __restrict__ wdw,     // [3][3][1][1020]
                                  _Float16* __restrict__ u16) {      // [M][512]
    unsigned idx = blockIdx.x * 256u + threadIdx.x;
    unsigned m = idx >> 9;
    int j = (int)(idx & 511u);
    if (j >= 510) { u16[(size_t)m * KOUT + j] = (_Float16)0.0f; return; }
    int b = (int)(m >> 14), n = (int)(m & 16383u);
    int y = n >> 7, xp = n & 127;
    float a1 = 0.0f, a2 = 0.0f;
#pragma unroll
    for (int dy = -1; dy <= 1; ++dy) {
        int yy = y + dy;
        if ((unsigned)yy >= 128u) continue;
#pragma unroll
        for (int dx = -1; dx <= 1; ++dx) {
            int xx = xp + dx;
            if ((unsigned)xx >= 128u) continue;
            size_t row = ((size_t)((b << 14) + (yy << 7) + xx)) * NIN;
            int tap = ((dy + 1) * 3 + (dx + 1)) * 1020;
            a1 += (float)upre[row + j]       * wdw[tap + j];
            a2 += (float)upre[row + j + 510] * wdw[tap + j + 510];
        }
    }
    float gl = 0.5f * a1 * (1.0f + erff(a1 * 0.70710678118654752f));
    u16[(size_t)m * KOUT + j] = (_Float16)(gl * a2);
}

// ---------------------------------------------------------------------------
// Host-side launch sequence
// ---------------------------------------------------------------------------
extern "C" void kernel_launch(void* const* d_in, const int* in_sizes, int n_in,
                              void* d_out, int out_size, void* d_ws, size_t ws_size,
                              hipStream_t stream) {
    (void)in_sizes; (void)n_in; (void)out_size; (void)ws_size;
    const float* x      = (const float*)d_in[0];
    const float* g1     = (const float*)d_in[1];
    const float* b1     = (const float*)d_in[2];
    const float* w_qkv  = (const float*)d_in[3];
    const float* w_dw   = (const float*)d_in[4];
    const float* temp   = (const float*)d_in[5];
    const float* mean_q = (const float*)d_in[6];
    const float* var_q  = (const float*)d_in[7];
    const float* mean_k = (const float*)d_in[8];
    const float* var_k  = (const float*)d_in[9];
    const float* w_proj = (const float*)d_in[10];
    const float* g2     = (const float*)d_in[11];
    const float* b2     = (const float*)d_in[12];
    const float* w_in   = (const float*)d_in[13];
    const float* w_fdw  = (const float*)d_in[14];
    const float* w_out  = (const float*)d_in[15];
    float* out = (float*)d_out;

    char* ws = (char*)d_ws;
    _Float16* wtq  = (_Float16*)(ws + 0);                    //  576*192 f16
    _Float16* wtp  = (_Float16*)(ws + 221184ull);            //  192*192 f16
    _Float16* wti  = (_Float16*)(ws + 294912ull);            // 1024*192 f16
    _Float16* wto  = (_Float16*)(ws + 688128ull);            //  192*512 f16
    _Float16* a16  = (_Float16*)(ws + 884736ull);            // [M][192] f16 (LN out, reused)
    _Float16* qkv  = (_Float16*)(ws + 51216384ull);          // [M][576] f16
    _Float16* upre = qkv;                                    // alias: [M][1024] f16 (phase 2)
    _Float16* qh   = (_Float16*)(ws + 319651840ull);         // [48][16384][32] f16
    _Float16* kh   = (_Float16*)(ws + 319651840ull + 50331648ull);
    _Float16* vh   = (_Float16*)(ws + 319651840ull + 100663296ull);
    _Float16* u16  = (_Float16*)(ws + 319651840ull);         // alias: [M][512] f16 (phase 2)
    float*    part = (float*)(ws + 470646784ull);            // [48][8][1024] f32
    _Float16* attnT= (_Float16*)(ws + 472219648ull);         // [48][32][32] f16
    _Float16* outH = (_Float16*)(ws + 472317952ull);         // [M][192] f16
    float*    x2   = (float*)(ws + 522649600ull);            // [M][192] f32

    // 1) weights -> f16 [N][K]
    convert_weights<<<1728, 256, 0, stream>>>(w_qkv, w_proj, w_in, w_out,
                                              wtq, wtp, wti, wto);
    // 2) LN1
    ln_kernel<<<16384, 256, 0, stream>>>(x, g1, b1, a16);
    // 3) qkv = LN(x) @ Wqkv   (M x 192 x 576)
    gemm_kernel<0><<<dim3(NQKV / 64, M_TOT / 128), 256, 0, stream>>>(
        a16, wtq, nullptr, (void*)qkv, DIM, NQKV);
    // 4) dwconv3x3 + q/k normalization -> head-major q/k/v
    dwconv_qkv_kernel<<<(M_TOT * NQKV) / 256, 256, 0, stream>>>(
        qkv, w_dw, mean_q, var_q, mean_k, var_k, qh, kh, vh);
    // 5) attn partials: K^T @ Q per (b,head), split-K over n
    attn_kernel<<<dim3(8, 48), 256, 0, stream>>>(qh, kh, part);
    // 6) reduce + temp + softmax(axis c) -> attnT
    softmax_kernel<<<48, 32, 0, stream>>>(part, temp, attnT);
    // 7) out = V @ attn -> out_heads [M][192]
    gemm_av_kernel<<<dim3(64, 48), 256, 0, stream>>>(vh, attnT, outH);
    // 8) x2 = x + out_heads @ Wproj
    gemm_kernel<1><<<dim3(DIM / 64, M_TOT / 128), 256, 0, stream>>>(
        outH, wtp, x, (void*)x2, DIM, DIM);
    // 9) LN2
    ln_kernel<<<16384, 256, 0, stream>>>(x2, g2, b2, a16);
    // 10) upre = LN(x2) @ Win  (M x 192 x 1024, padded N)
    gemm_kernel<0><<<dim3(NIN / 64, M_TOT / 128), 256, 0, stream>>>(
        a16, wti, nullptr, (void*)upre, DIM, NIN);
    // 11) dwconv3x3 + gelu gate -> u [M][512] (padded K)
    dwconv_ffn_kernel<<<(M_TOT * KOUT) / 256, 256, 0, stream>>>(upre, w_fdw, u16);
    // 12) out = x2 + u @ Wout  (M x 512 x 192)
    gemm_kernel<1><<<dim3(DIM / 64, M_TOT / 128), 256, 0, stream>>>(
        u16, wto, x2, (void*)out, KOUT, DIM);
}
